// IsingLayer_10582799417798
// MI455X (gfx1250) — matrix operand order repaired and checked by reading, not compiled, not run on hardware
//
#include <hip/hip_runtime.h>
#include <hip/hip_bf16.h>
#include <stdint.h>

typedef __attribute__((ext_vector_type(16))) _Float16 v16h;
typedef __attribute__((ext_vector_type(8)))  _Float16 v8h;
typedef __attribute__((ext_vector_type(8)))  float    v8f;

#define NSPIN 256
#define NSWEEP 200
#define WAVES_PER_BLK 2
#define ROWS_PER_WAVE 16
#define ROWS_PER_BLK (WAVES_PER_BLK * ROWS_PER_WAVE)   // 32
#define THREADS (WAVES_PER_BLK * 32)                   // 64

// Dynamic LDS layout (bytes), 160 KB total -> two blocks fit per 320 KB WGP:
//   Jh  : [0,      131072)  _Float16[256*256]   symmetric couplings (f16)
//   S   : [131072, 147456)  _Float16[32*256]    spin state (per-wave private rows)
//   ThT : [147456, 163840)  _Float16[2][256*16] thetas, C-fragment transpose (f16)
#define JH_BYTES   (NSPIN * NSPIN * 2)                    // 131072
#define S_BYTES    (ROWS_PER_BLK * NSPIN * 2)             // 16384
#define THT_BYTES  (WAVES_PER_BLK * NSPIN * 16 * 2)       // 16384
#define SMEM_BYTES (JH_BYTES + S_BYTES + THT_BYTES)       // 163840

__device__ __forceinline__ uint32_t ih(uint32_t x) {
  x ^= x >> 16; x *= 0x7feb352dU;
  x ^= x >> 15; x *= 0x846ca68bU;
  x ^= x >> 16;
  return x;
}

__device__ __forceinline__ v16h load_bfrag(const _Float16* bp) {
  union { v16h v; v8h h[2]; } u;
  u.h[0] = *(const v8h*)(bp);
  u.h[1] = *(const v8h*)(bp + 8);
  return u.v;
}

__global__ __launch_bounds__(THREADS)
void ising_anneal_kernel(const float* __restrict__ thetas,
                         const float* __restrict__ gam,
                         float* __restrict__ out)
{
  extern __shared__ char smem[];
  _Float16* Jh  = (_Float16*)smem;
  _Float16* S   = (_Float16*)(smem + JH_BYTES);
  _Float16* ThT = (_Float16*)(smem + JH_BYTES + S_BYTES);

  const int tid     = threadIdx.x;
  const int lane    = tid & 31;
  const int wave    = tid >> 5;
  const int blkRow0 = blockIdx.x * ROWS_PER_BLK;

  // ---- build Jsym = triu(gamma,1) + triu(gamma,1)^T as f16 in LDS ----
  for (int e = tid; e < NSPIN * NSPIN; e += THREADS) {
    int i = e >> 8, j = e & 255;
    float v = 0.f;
    if (i < j)      v = gam[i * NSPIN + j];
    else if (i > j) v = gam[j * NSPIN + i];
    Jh[e] = (_Float16)v;
  }
  // ---- transposed thetas (f16, per wave) + deterministic initial spins ----
  for (int e = tid; e < ROWS_PER_BLK * NSPIN; e += THREADS) {
    int r = e >> 8;          // row within block
    int c = e & 255;         // spin index
    int w = r >> 4, m = r & 15;
    ThT[w * (NSPIN * 16) + c * 16 + m] = (_Float16)thetas[(blkRow0 + r) * NSPIN + c];
    uint32_t h = ih(((uint32_t)(blkRow0 + r) * 977u) ^ ((uint32_t)c * 131u) ^ 0x1234567u);
    S[r * NSPIN + c] = (h & 1u) ? (_Float16)1.0f : (_Float16)(-1.0f);
  }
  __syncthreads();

  _Float16* Sw = S + wave * ROWS_PER_WAVE * NSPIN;
  _Float16* Tw = ThT + wave * (NSPIN * 16);

  const int mA       = lane & 15;              // A-frag: M = lane%16
  const int aoff     = (lane < 16) ? 0 : 16;   // A-frag K sub-offset
  const int boff     = (lane < 16) ? 0 : 16;   // B-frag K sub-offset
  const int nLoc     = lane & 15;              // C/B-frag: N = lane%16
  const int mC0      = (lane < 16) ? 0 : 8;    // C-frag row base
  const int waveRow0 = blkRow0 + wave * ROWS_PER_WAVE;

  #pragma unroll 1
  for (int t = 0; t < NSWEEP; ++t) {
    // geomspace(0.1, 5.0, 200): beta = 0.1 * exp(t * ln(50)/199)
    float beta = 0.1f * __expf((float)t * 0.019658413f);
    float twoB = 2.0f * beta;

    // Snapshot spins as WMMA A-fragments (synchronous-update semantics).
    v16h afrag[8];
    #pragma unroll
    for (int kt = 0; kt < 8; ++kt) {
      const _Float16* p = Sw + mA * NSPIN + kt * 32 + aoff;
      union { v16h v; v8h h[2]; } u;
      u.h[0] = *(const v8h*)(p);        // K block +0
      u.h[1] = *(const v8h*)(p + 8);    // K block +8
      afrag[kt] = u.v;
    }

    // Four N-tiles per iteration: 4 independent WMMA chains share A-frags.
    // All 4 B-fragments of a k-step are loaded into simultaneously-live regs
    // BEFORE any WMMA of that step, forcing a partial-wait ladder
    // (dscnt<=6,4,2,0) so LDS latency is covered by in-flight WMMAs.
    #pragma unroll 1
    for (int n = 0; n < 16; n += 4) {
      v8f acc[4];
      #pragma unroll
      for (int q = 0; q < 4; ++q) {
        const _Float16* tp = Tw + ((n + q) * 16 + nLoc) * 16 + mC0;
        v8h th = *(const v8h*)tp;
        #pragma unroll
        for (int v = 0; v < 8; ++v) acc[q][v] = (float)th[v];
      }

      const _Float16* bp = Jh + (n * 16 + nLoc) * NSPIN + boff;

      #pragma unroll
      for (int kt = 0; kt < 8; ++kt) {
        v16h b[4];
        #pragma unroll
        for (int q = 0; q < 4; ++q)
          b[q] = load_bfrag(bp + q * (16 * NSPIN) + kt * 32);
        #pragma unroll
        for (int q = 0; q < 4; ++q)
          acc[q] = __builtin_amdgcn_wmma_f32_16x16x32_f16(
              false, afrag[kt], false, b[q], (short)0, acc[q], false, false);
      }

      // Branchless Metropolis update for the four 16x16 tiles (8 spins/lane).
      #pragma unroll
      for (int q = 0; q < 4; ++q) {
        int col = (n + q) * 16 + nLoc;
        #pragma unroll
        for (int v = 0; v < 8; ++v) {
          int m = mC0 + v;
          _Float16* sp = Sw + m * NSPIN + col;
          float sv  = (float)(*sp);
          float loc = acc[q][v];
          // accept prob: exp(-beta*dE), dE = -2*s*local  ->  exp(2*beta*s*local)
          float thr = __expf(twoB * sv * loc);
          uint32_t key = ((uint32_t)(waveRow0 + m) << 16) |
                         ((uint32_t)col << 8) | (uint32_t)t;
          uint32_t h  = ih(key ^ 0xA511E9B3u);
          float    uf = (float)(h >> 8) * (1.0f / 16777216.0f);
          int flip = (int)(h & 1u) & (int)(uf < thr);
          *sp = (_Float16)(flip ? -sv : sv);
        }
      }
    }
  }

  // ---- fused energy: E_b = 0.5 * sum_i s_i * (local_i + theta_i) ----
  {
    v16h afrag[8];
    #pragma unroll
    for (int kt = 0; kt < 8; ++kt) {
      const _Float16* p = Sw + mA * NSPIN + kt * 32 + aoff;
      union { v16h v; v8h h[2]; } u;
      u.h[0] = *(const v8h*)(p);
      u.h[1] = *(const v8h*)(p + 8);
      afrag[kt] = u.v;
    }

    float eacc[8];
    #pragma unroll
    for (int v = 0; v < 8; ++v) eacc[v] = 0.f;

    #pragma unroll 1
    for (int n = 0; n < 16; ++n) {
      const _Float16* tp = Tw + (n * 16 + nLoc) * 16 + mC0;
      v8h th8 = *(const v8h*)tp;
      v8f tf, acc;
      #pragma unroll
      for (int v = 0; v < 8; ++v) { tf[v] = (float)th8[v]; acc[v] = tf[v]; }

      const _Float16* bp = Jh + (n * 16 + nLoc) * NSPIN + boff;
      #pragma unroll
      for (int kt = 0; kt < 8; ++kt) {
        v16h b = load_bfrag(bp + kt * 32);
        acc = __builtin_amdgcn_wmma_f32_16x16x32_f16(
            false, afrag[kt], false, b, (short)0, acc, false, false);
      }

      int col = n * 16 + nLoc;
      #pragma unroll
      for (int v = 0; v < 8; ++v) {
        int m = mC0 + v;
        float sv = (float)(Sw[m * NSPIN + col]);
        eacc[v] += sv * 0.5f * (acc[v] + tf[v]);
      }
    }

    // reduce across the 16 lanes of each half-wave (rows 0-7 in lanes 0-15,
    // rows 8-15 in lanes 16-31)
    #pragma unroll
    for (int v = 0; v < 8; ++v) {
      #pragma unroll
      for (int off = 1; off < 16; off <<= 1)
        eacc[v] += __shfl_xor(eacc[v], off, 32);
    }
    if (nLoc == 0) {  // lanes 0 and 16
      #pragma unroll
      for (int v = 0; v < 8; ++v)
        out[waveRow0 + mC0 + v] = eacc[v];
    }
  }
}

extern "C" void kernel_launch(void* const* d_in, const int* in_sizes, int n_in,
                              void* d_out, int out_size, void* d_ws, size_t ws_size,
                              hipStream_t stream) {
  (void)in_sizes; (void)n_in; (void)out_size; (void)d_ws; (void)ws_size;
  const float* thetas = (const float*)d_in[0];  // [4096, 256] f32
  const float* gam    = (const float*)d_in[1];  // [256, 256]  f32
  float* out          = (float*)d_out;          // [4096]      f32

  const int B = 4096;
  dim3 grid(B / ROWS_PER_BLK);   // 128 blocks -> spread LDS traffic over WGPs
  dim3 block(THREADS);           // 64 threads = 2 waves (wave32)
  ising_anneal_kernel<<<grid, block, SMEM_BYTES, stream>>>(thetas, gam, out);
}
